// TimeMix_40106404610577
// MI455X (gfx1250) — matrix operand order repaired and checked by reading, compiled, not run on hardware
//
#include <hip/hip_runtime.h>
#include <hip/hip_bf16.h>

// ---------------- problem constants (match reference) ----------------
constexpr int B_ = 4, T_ = 512, C_ = 2048, H_ = 32, N_ = 64;
constexpr int DMIX = 32, DDEC = 64;
constexpr float EPS_ = 1e-5f * 64.0f;

typedef __bf16 bf16_t;
typedef __attribute__((ext_vector_type(16))) __bf16 v16bf;
typedef __attribute__((ext_vector_type(8)))  float  v8f;
typedef unsigned int uint32;

// ---- CDNA5 async global->LDS copy (ASYNCcnt-tracked) ----
__device__ __forceinline__ void async_copy_b128(unsigned lds_off, const void* gaddr) {
  asm volatile("global_load_async_to_lds_b128 %0, %1, off"
               : : "v"(lds_off), "v"(gaddr) : "memory");
}
__device__ __forceinline__ void wait_async0() {
  asm volatile("s_wait_asynccnt 0x0" : : : "memory");
}

// ======================================================================
// Kernel 0: fp32 weight -> bf16 TRANSPOSED weight  (W[K][N] -> WT[N][K])
// 64x64 tile per block, 256 threads
// ======================================================================
__global__ __launch_bounds__(256) void cvt_transpose_bf16(
    const float* __restrict__ W, bf16_t* __restrict__ WT) {
  __shared__ bf16_t tile[64][64 + 4];
  int k0 = blockIdx.x * 64;
  int n0 = blockIdx.y * 64;
  int tid = threadIdx.x;
#pragma unroll
  for (int rep = 0; rep < 16; ++rep) {
    int i = tid + rep * 256;          // 0..4095
    int k = i >> 6, n = i & 63;       // coalesced along n
    tile[n][k] = (bf16_t)W[(size_t)(k0 + k) * C_ + n0 + n];
  }
  __syncthreads();
#pragma unroll
  for (int rep = 0; rep < 8; ++rep) {
    int i = tid + rep * 256;          // 0..2047 (pairs)
    int n = i >> 5, k = (i & 31) * 2;
    union { bf16_t h[2]; uint32 u; } p;
    p.h[0] = tile[n][k]; p.h[1] = tile[n][k + 1];
    *(uint32*)(WT + (size_t)(n0 + n) * C_ + k0 + k) = p.u;
  }
}

// ======================================================================
// Kernel 1: token shift + 5-way low-rank mix
// outputs: xw (f32, feeds decay MLP) and xk,xv,xr,xg (bf16, feed WMMA GEMMs)
// ======================================================================
__global__ __launch_bounds__(256) void mix_prep(
    const float* __restrict__ x, const float* __restrict__ shift_state,
    const float* __restrict__ maa_x, const float* __restrict__ maa_w,
    const float* __restrict__ maa_k, const float* __restrict__ maa_v,
    const float* __restrict__ maa_r, const float* __restrict__ maa_g,
    const float* __restrict__ maa_w1 /*C x 160*/,
    const float* __restrict__ maa_w2 /*5 x 32 x C*/,
    float* __restrict__ xw, bf16_t* __restrict__ xk, bf16_t* __restrict__ xv,
    bf16_t* __restrict__ xr, bf16_t* __restrict__ xg) {
  __shared__ float s_mix[C_];
  __shared__ float s_dx[C_];
  __shared__ float s_xxx[5 * DMIX];
  int bt = blockIdx.x;
  int b = bt / T_, t = bt % T_;
  int tid = threadIdx.x;
  const float* xrow = x + (size_t)bt * C_;
  const float* prev = (t == 0) ? (shift_state + (size_t)b * C_)
                               : (x + (size_t)(bt - 1) * C_);
  for (int c = tid; c < C_; c += 256) {
    float dxv = prev[c] - xrow[c];
    s_dx[c]  = dxv;
    s_mix[c] = xrow[c] + dxv * maa_x[c];
  }
  __syncthreads();
  for (int o = tid; o < 5 * DMIX; o += 256) {
    float acc = 0.f;
    for (int c = 0; c < C_; ++c) acc += s_mix[c] * maa_w1[(size_t)c * (5 * DMIX) + o];
    s_xxx[o] = tanhf(acc);
  }
  __syncthreads();
  for (int c = tid; c < C_; c += 256) {
    float m[5];
#pragma unroll
    for (int f = 0; f < 5; ++f) {
      float acc = 0.f;
#pragma unroll
      for (int d = 0; d < DMIX; ++d)
        acc += s_xxx[f * DMIX + d] * maa_w2[(size_t)(f * DMIX + d) * C_ + c];
      m[f] = acc;
    }
    float xv0 = xrow[c], dxv = s_dx[c];
    size_t o = (size_t)bt * C_ + c;
    xw[o] = xv0 + dxv * (maa_w[c] + m[0]);
    xk[o] = (bf16_t)(xv0 + dxv * (maa_k[c] + m[1]));
    xv[o] = (bf16_t)(xv0 + dxv * (maa_v[c] + m[2]));
    xr[o] = (bf16_t)(xv0 + dxv * (maa_r[c] + m[3]));
    xg[o] = (bf16_t)(xv0 + dxv * (maa_g[c] + m[4]));
  }
}

// ======================================================================
// Kernel 2: decay = exp(-exp(time_decay + tanh(xw@w1)@w2))
// ======================================================================
__global__ __launch_bounds__(256) void decay_prep(
    const float* __restrict__ xw, const float* __restrict__ w1 /*C x 64*/,
    const float* __restrict__ w2 /*64 x C*/, const float* __restrict__ time_decay,
    float* __restrict__ decay_out) {
  __shared__ float s_row[C_];
  __shared__ float s_h[DDEC];
  int bt = blockIdx.x, tid = threadIdx.x;
  const float* row = xw + (size_t)bt * C_;
  for (int c = tid; c < C_; c += 256) s_row[c] = row[c];
  __syncthreads();
  for (int o = tid; o < DDEC; o += 256) {
    float acc = 0.f;
    for (int c = 0; c < C_; ++c) acc += s_row[c] * w1[(size_t)c * DDEC + o];
    s_h[o] = tanhf(acc);
  }
  __syncthreads();
  for (int c = tid; c < C_; c += 256) {
    float acc = time_decay[c];
#pragma unroll
    for (int d = 0; d < DDEC; ++d) acc += s_h[d] * w2[(size_t)d * C_ + c];
    decay_out[(size_t)bt * C_ + c] = __expf(-__expf(acc));
  }
}

// ======================================================================
// Kernel 3: bf16 WMMA GEMM  C[MxN] = A[MxK] * WT[NxK]^T  (both bf16)
// 256 thr (8 waves), tile 128x128x32, double-buffered async global->LDS.
// Each wave: 64x32 output = 4x2 WMMA tiles.
// ======================================================================
#define BM 128
#define BN 128
#define BK 32
#define LDK (BK + 8)   // 40 bf16 = 80B row stride

__global__ __launch_bounds__(256) void gemm_bf16_wmma(
    const bf16_t* __restrict__ A, const bf16_t* __restrict__ WT,
    float* __restrict__ Cout, int M, int N, int Kd) {
  __shared__ __align__(16) bf16_t As[2][BM][LDK];
  __shared__ __align__(16) bf16_t Bs[2][BN][LDK];

  const int tid  = threadIdx.x;
  const int lane = tid & 31;
  const int wid  = tid >> 5;
  const int half = lane >> 4;
  const int lm   = lane & 15;
  const int bm   = blockIdx.y * BM;
  const int bn   = blockIdx.x * BN;
  const int wm   = (wid & 1) * 64;
  const int wn   = (wid >> 1) * 32;

  v8f acc[4][2];
#pragma unroll
  for (int i = 0; i < 4; ++i)
#pragma unroll
    for (int j = 0; j < 2; ++j) acc[i][j] = (v8f)(0.0f);

  auto stage = [&](int kt, int buf) {
    unsigned ab = (unsigned)(size_t)(&As[buf][0][0]);
    unsigned bb = (unsigned)(size_t)(&Bs[buf][0][0]);
#pragma unroll
    for (int rep = 0; rep < 2; ++rep) {
      int slot = tid + rep * 256;       // 512 slots of 8 bf16
      int r  = slot >> 2;               // 0..127
      int kk = (slot & 3) * 8;          // 0,8,16,24
      async_copy_b128(ab + (unsigned)(r * LDK + kk) * 2,
                      A + (size_t)(bm + r) * Kd + kt + kk);
      async_copy_b128(bb + (unsigned)(r * LDK + kk) * 2,
                      WT + (size_t)(bn + r) * Kd + kt + kk);
    }
  };

  auto compute = [&](int buf) {
    v16bf afrag[4], bfrag[2];
#pragma unroll
    for (int sm = 0; sm < 4; ++sm) {
      const bf16_t* p = &As[buf][wm + sm * 16 + lm][half * 8];
      union { uint4 q[2]; v16bf v; } f;
      f.q[0] = *(const uint4*)p;          // K: half*8 .. +7   (VGPR 0-3)
      f.q[1] = *(const uint4*)(p + 16);   // K: 16+half*8 .. +7 (VGPR 4-7)
      afrag[sm] = f.v;
    }
#pragma unroll
    for (int sn = 0; sn < 2; ++sn) {
      const bf16_t* p = &Bs[buf][wn + sn * 16 + lm][half * 8];
      union { uint4 q[2]; v16bf v; } f;
      f.q[0] = *(const uint4*)p;          // K: half*8 .. +7
      f.q[1] = *(const uint4*)(p + 16);   // K: 16+half*8 .. +7
      bfrag[sn] = f.v;
    }
#pragma unroll
    for (int sm = 0; sm < 4; ++sm)
#pragma unroll
      for (int sn = 0; sn < 2; ++sn)
        acc[sm][sn] = __builtin_amdgcn_wmma_f32_16x16x32_bf16(
            false, afrag[sm], false, bfrag[sn], (short)0, acc[sm][sn], false, false);
  };

  stage(0, 0);
  wait_async0();
  __syncthreads();
  int cur = 0;
  for (int kt = 0; kt < Kd; kt += BK) {
    if (kt + BK < Kd) stage(kt + BK, cur ^ 1);  // overlap HBM->LDS with WMMA
    compute(cur);
    wait_async0();
    __syncthreads();
    cur ^= 1;
  }

#pragma unroll
  for (int sm = 0; sm < 4; ++sm)
#pragma unroll
    for (int sn = 0; sn < 2; ++sn) {
      int col = bn + wn + sn * 16 + lm;
#pragma unroll
      for (int r = 0; r < 8; ++r) {
        int row = bm + wm + sm * 16 + r + half * 8;
        Cout[(size_t)row * N + col] = acc[sm][sn][r];
      }
    }
}

// ======================================================================
// Kernel 4: WKV6 sequential scan. one block per (b,h), 64 threads.
// ======================================================================
__global__ __launch_bounds__(64) void wkv6_scan(
    const float* __restrict__ r, const float* __restrict__ k,
    const float* __restrict__ v, const float* __restrict__ decay,
    const float* __restrict__ u, const float* __restrict__ s0,
    float* __restrict__ y) {
  int bh = blockIdx.x;
  int b = bh / H_, h = bh % H_;
  int j = threadIdx.x;
  __shared__ float rs[N_], ks[N_], ds[N_], us[N_];
  us[j] = u[h * N_ + j];

  float s[N_];
  const float* s0p = s0 + (size_t)bh * N_ * N_;
#pragma unroll
  for (int i = 0; i < N_; ++i) s[i] = s0p[i * N_ + j];

  for (int t = 0; t < T_; ++t) {
    size_t base = ((size_t)(b * T_ + t) * H_ + h) * N_;
    rs[j] = r[base + j];
    ks[j] = k[base + j];
    ds[j] = decay[base + j];
    __syncthreads();
    float vj = v[base + j];
    float yj = 0.f;
#pragma unroll
    for (int i = 0; i < N_; ++i) {
      float kv = ks[i] * vj;
      yj  += rs[i] * (s[i] + us[i] * kv);
      s[i] = ds[i] * s[i] + kv;
    }
    y[base + j] = yj;
    __syncthreads();
  }
}

// ======================================================================
// Kernel 5: per-head groupnorm + ln affine + SiLU gate -> yg (bf16)
// ======================================================================
__global__ __launch_bounds__(64) void norm_gate(
    const float* __restrict__ y, const float* __restrict__ gpre,
    const float* __restrict__ ln_g, const float* __restrict__ ln_b,
    bf16_t* __restrict__ yg) {
  int idx = blockIdx.x;           // (b*T + t)*H + h
  int h = idx % H_;
  int bt = idx / H_;
  int j = threadIdx.x;
  float vy = y[(size_t)idx * N_ + j];
  __shared__ float red[N_], red2[N_];
  red[j] = vy; red2[j] = vy * vy;
  __syncthreads();
#pragma unroll
  for (int s = N_ / 2; s > 0; s >>= 1) {
    if (j < s) { red[j] += red[j + s]; red2[j] += red2[j + s]; }
    __syncthreads();
  }
  float mean = red[0] * (1.0f / N_);
  float var  = red2[0] * (1.0f / N_) - mean * mean;
  float yn   = (vy - mean) * rsqrtf(var + EPS_);
  int c = h * N_ + j;
  float g = gpre[(size_t)bt * C_ + c];
  float silu = g / (1.0f + __expf(-g));
  yg[(size_t)bt * C_ + c] = (bf16_t)((yn * ln_g[c] + ln_b[c]) * silu);
}

// ======================================================================
// Kernel 6: tail outputs: shift_out = x[:, -1]; wkv_state passthrough
// ======================================================================
__global__ __launch_bounds__(256) void tail_copy(
    const float* __restrict__ x, const float* __restrict__ state,
    float* __restrict__ out_tail, int nshift, int nstate) {
  int i = blockIdx.x * 256 + threadIdx.x;
  if (i < nshift) {
    int b = i / C_, c = i % C_;
    out_tail[i] = x[((size_t)b * T_ + (T_ - 1)) * C_ + c];
  } else if (i < nshift + nstate) {
    out_tail[i] = state[i - nshift];
  }
}

// ======================================================================
extern "C" void kernel_launch(void* const* d_in, const int* in_sizes, int n_in,
                              void* d_out, int out_size, void* d_ws, size_t ws_size,
                              hipStream_t stream) {
  (void)in_sizes; (void)n_in; (void)out_size; (void)ws_size;
  const float* x          = (const float*)d_in[0];
  const float* shift_st   = (const float*)d_in[1];
  const float* wkv_st     = (const float*)d_in[2];
  const float* maa_x      = (const float*)d_in[3];
  const float* maa_w      = (const float*)d_in[4];
  const float* maa_k      = (const float*)d_in[5];
  const float* maa_v      = (const float*)d_in[6];
  const float* maa_r      = (const float*)d_in[7];
  const float* maa_g      = (const float*)d_in[8];
  const float* maa_w1     = (const float*)d_in[9];
  const float* maa_w2     = (const float*)d_in[10];
  const float* time_decay = (const float*)d_in[11];
  const float* decay_w1   = (const float*)d_in[12];
  const float* decay_w2   = (const float*)d_in[13];
  const float* time_faaaa = (const float*)d_in[14];
  const float* Wr = (const float*)d_in[15];
  const float* Wk = (const float*)d_in[16];
  const float* Wv = (const float*)d_in[17];
  const float* Wg = (const float*)d_in[18];
  const float* Wo = (const float*)d_in[19];
  const float* ln_g = (const float*)d_in[20];
  const float* ln_b = (const float*)d_in[21];
  float* out = (float*)d_out;

  const size_t BTC = (size_t)B_ * T_ * C_;
  char* wsb = (char*)d_ws;
  size_t off = 0;
  auto take = [&](size_t bytes) -> char* {
    char* p = wsb + off;
    off += (bytes + 255) & ~(size_t)255;
    return p;
  };
  const size_t wbytes = (size_t)C_ * C_ * sizeof(bf16_t);
  bf16_t* WrT = (bf16_t*)take(wbytes);
  bf16_t* WkT = (bf16_t*)take(wbytes);
  bf16_t* WvT = (bf16_t*)take(wbytes);
  bf16_t* WgT = (bf16_t*)take(wbytes);
  bf16_t* WoT = (bf16_t*)take(wbytes);
  float*  xw  = (float*)take(BTC * 4);
  bf16_t* xkb = (bf16_t*)take(BTC * 2);
  bf16_t* xvb = (bf16_t*)take(BTC * 2);
  bf16_t* xrb = (bf16_t*)take(BTC * 2);
  bf16_t* xgb = (bf16_t*)take(BTC * 2);
  float* rb  = (float*)take(BTC * 4);
  float* kb  = (float*)take(BTC * 4);
  float* vb  = (float*)take(BTC * 4);
  float* gp  = (float*)take(BTC * 4);
  float* dec = (float*)take(BTC * 4);
  float* yb  = (float*)take(BTC * 4);
  bf16_t* ygb = (bf16_t*)take(BTC * 2);

  // 0) weights -> bf16, transposed (so GEMM B tiles are k-contiguous)
  {
    dim3 g(C_ / 64, C_ / 64), blk(256);
    cvt_transpose_bf16<<<g, blk, 0, stream>>>(Wr, WrT);
    cvt_transpose_bf16<<<g, blk, 0, stream>>>(Wk, WkT);
    cvt_transpose_bf16<<<g, blk, 0, stream>>>(Wv, WvT);
    cvt_transpose_bf16<<<g, blk, 0, stream>>>(Wg, WgT);
    cvt_transpose_bf16<<<g, blk, 0, stream>>>(Wo, WoT);
  }
  // 1) token shift + low-rank mixing
  mix_prep<<<dim3(B_ * T_), dim3(256), 0, stream>>>(
      x, shift_st, maa_x, maa_w, maa_k, maa_v, maa_r, maa_g,
      maa_w1, maa_w2, xw, xkb, xvb, xrb, xgb);
  // 2) decay
  decay_prep<<<dim3(B_ * T_), dim3(256), 0, stream>>>(
      xw, decay_w1, decay_w2, time_decay, dec);
  // 3) big WMMA GEMMs: r,k,v,g
  {
    dim3 g(C_ / BN, (B_ * T_) / BM), blk(256);
    gemm_bf16_wmma<<<g, blk, 0, stream>>>(xrb, WrT, rb, B_ * T_, C_, C_);
    gemm_bf16_wmma<<<g, blk, 0, stream>>>(xkb, WkT, kb, B_ * T_, C_, C_);
    gemm_bf16_wmma<<<g, blk, 0, stream>>>(xvb, WvT, vb, B_ * T_, C_, C_);
    gemm_bf16_wmma<<<g, blk, 0, stream>>>(xgb, WgT, gp, B_ * T_, C_, C_);
  }
  // 4) WKV scan
  wkv6_scan<<<dim3(B_ * H_), dim3(N_), 0, stream>>>(
      rb, kb, vb, dec, time_faaaa, wkv_st, yb);
  // 5) groupnorm + gate
  norm_gate<<<dim3(B_ * T_ * H_), dim3(N_), 0, stream>>>(yb, gp, ln_g, ln_b, ygb);
  // 6) output projection -> d_out
  {
    dim3 g(C_ / BN, (B_ * T_) / BM), blk(256);
    gemm_bf16_wmma<<<g, blk, 0, stream>>>(ygb, WoT, out, B_ * T_, C_, C_);
  }
  // 7) tail: shift_out + passthrough state
  {
    int nshift = B_ * C_, nstate = B_ * H_ * N_ * N_;
    int ntot = nshift + nstate;
    tail_copy<<<dim3((ntot + 255) / 256), dim3(256), 0, stream>>>(
        x, wkv_st, out + BTC, nshift, nstate);
  }
}
